// AttNet_60730837566330
// MI455X (gfx1250) — matrix operand order repaired and checked by reading, not compiled, not run on hardware
//
#include <hip/hip_runtime.h>

typedef __attribute__((ext_vector_type(16))) _Float16 v16h;
typedef __attribute__((ext_vector_type(8)))  _Float16 v8h;
typedef __attribute__((ext_vector_type(8)))  float    v8f;

#define BN_EPS 1e-5f

// ---------------------------------------------------------------------------
// zero only the 1-cell border of each padded (Hp x Wp) channel plane
// perim = 2*Wp + 2*H cells per channel ; n = B*C*perim
// ---------------------------------------------------------------------------
__global__ void pad_border_kernel(float* __restrict__ P, int Wp, int H, int perim, int n)
{
  int idx = blockIdx.x * blockDim.x + threadIdx.x;
  if (idx >= n) return;
  int ch = idx / perim;            // flattened (b*C + c)
  int i  = idx - ch * perim;
  int padN = (H + 2) * Wp;
  int sp;
  if (i < Wp)            sp = i;                              // top row
  else if (i < 2 * Wp)   sp = (H + 1) * Wp + (i - Wp);        // bottom row
  else {
    int j = i - 2 * Wp;                                       // left/right cols
    int row = 1 + (j >> 1);
    sp = row * Wp + ((j & 1) ? (Wp - 1) : 0);
  }
  P[(size_t)ch * padN + sp] = 0.0f;
}

// ---------------------------------------------------------------------------
// activation f32 (B,Cin,HW) -> f16 K-contiguous (B*HW, Cinp), zero K-padding
// one thread per (b,hw); the k<Cin bound is uniform across the wave
// ---------------------------------------------------------------------------
__global__ void cvt_x_kernel(const float* __restrict__ X, _Float16* __restrict__ Xh,
                             int Cin, int Cinp, int HW, int n /* = B*HW */)
{
  int idx = blockIdx.x * blockDim.x + threadIdx.x;
  if (idx >= n) return;
  int b = idx / HW, hw = idx - b * HW;
  const float* xs = X + (size_t)b * Cin * HW + hw;
  _Float16* xd = Xh + (size_t)idx * Cinp;
  for (int k = 0; k < Cin; ++k)  xd[k] = (_Float16)xs[(size_t)k * HW];
  for (int k = Cin; k < Cinp; ++k) xd[k] = (_Float16)0.0f;
}

// weights f32 (Cout,Cin) -> f16 (Cout,Cinp), zero K-padding (tiny kernel)
__global__ void cvt_w_kernel(const float* __restrict__ Wt, _Float16* __restrict__ Wh,
                             int Cin, int Cinp, int n /* = Cout*Cinp */)
{
  int idx = blockIdx.x * blockDim.x + threadIdx.x;
  if (idx >= n) return;
  int co = idx / Cinp, k = idx - co * Cinp;
  Wh[idx] = (k < Cin) ? (_Float16)Wt[(size_t)co * Cin + k] : (_Float16)0.0f;
}

// ---------------------------------------------------------------------------
// 1x1 conv as GEMM: Y[b,co,n] = sum_ci W[co,ci] * X[b,ci,n]
// One wave32 computes a 16(Cout) x 64(spatial) f32 tile: A fragment loaded
// once per 32-K step (2x b128), 4 WMMAs reuse it against 4 B fragments
// (2x b128 each, K-contiguous).  pad==1 scatters into zero-bordered layout.
// ---------------------------------------------------------------------------
__global__ void conv1x1_wmma_kernel(const _Float16* __restrict__ Wh,  // (Cout,Cinp)
                                    const _Float16* __restrict__ Xh,  // (B*HW,Cinp)
                                    float* __restrict__ Y,
                                    int Cinp, int Cout, int HW,
                                    int Wimg, int pad, int Wp, int padN,
                                    int nWaves)
{
  int wave = (blockIdx.x * blockDim.x + threadIdx.x) >> 5;
  if (wave >= nWaves) return;                 // wave-uniform exit, EXEC stays all-1
  int lane = threadIdx.x & 31;
  int m  = lane & 15;                         // A row / B,C,D column index
  int hi = lane >> 4;                         // lane-half selector

  int tilesN = HW >> 6;                       // 64 spatial positions per wave
  int tilesM = Cout >> 4;
  int perB = tilesM * tilesN;
  int b  = wave / perB;
  int r  = wave - b * perB;
  int tm = r / tilesN;
  int tn = r - tm * tilesN;

  int co0 = tm << 4;
  int n0  = tn << 6;

  // A 16x32 f16 layout: lane half 'hi' holds K = 16*hi + {0..15}
  // B 32x16 f16 layout: lane half 'hi' holds K = 16*hi + {0..15}, N = lane&15
  const _Float16* arow = Wh + (size_t)(co0 + m) * Cinp + hi * 16;
  const _Float16* brow = Xh + ((size_t)b * HW + n0 + m) * Cinp + hi * 16;

  v8f acc[4] = {v8f{}, v8f{}, v8f{}, v8f{}};
  for (int k0 = 0; k0 < Cinp; k0 += 32) {
    v8h alo = *(const v8h*)(arow + k0);
    v8h ahi = *(const v8h*)(arow + k0 + 8);
    v16h afrag = __builtin_shufflevector(alo, ahi,
        0, 1, 2, 3, 4, 5, 6, 7, 8, 9, 10, 11, 12, 13, 14, 15);
#pragma unroll
    for (int nt = 0; nt < 4; ++nt) {
      const _Float16* bp = brow + (size_t)(nt * 16) * Cinp + k0;
      v8h blo = *(const v8h*)(bp);
      v8h bhi = *(const v8h*)(bp + 8);
      v16h bfrag = __builtin_shufflevector(blo, bhi,
          0, 1, 2, 3, 4, 5, 6, 7, 8, 9, 10, 11, 12, 13, 14, 15);
      acc[nt] = __builtin_amdgcn_wmma_f32_16x16x32_f16(false, afrag, false, bfrag,
                                                       (short)0, acc[nt], false, false);
    }
  }

  // C/D layout: lane column N = lane&15; VGPR r holds M = r + 8*hi
#pragma unroll
  for (int nt = 0; nt < 4; ++nt) {
    int n = n0 + nt * 16 + m;
    int sp;
    size_t chanN;
    if (pad) { int h = n / Wimg, w = n - h * Wimg; sp = (h + 1) * Wp + (w + 1); chanN = (size_t)padN; }
    else     { sp = n; chanN = (size_t)HW; }
    float* Yb = Y + (size_t)b * Cout * chanN;
#pragma unroll
    for (int rr = 0; rr < 8; ++rr)
      Yb[(size_t)(co0 + rr + hi * 8) * chanN + sp] = acc[nt][rr];
  }
}

// ---------------------------------------------------------------------------
// 3x3-window attention per pixel:
//   s[kh,kw] = sum_c (K[c,h+kh,w+kw] + bias(c,kh,kw)) * q[c]
//   attn = softmax(s); out[c] = sum attn * V[c, window]
// bias = rh[c,kh] for c < C/2, rw[c-C/2,kw] otherwise.
// ---------------------------------------------------------------------------
__global__ void attention_kernel(const float* __restrict__ Q,    // (B,C,HW)
                                 const float* __restrict__ KP,   // (B,C,Hp,Wp)
                                 const float* __restrict__ VP,   // (B,C,Hp,Wp)
                                 const float* __restrict__ rh,   // (C/2,3)
                                 const float* __restrict__ rw,   // (C/2,3)
                                 float* __restrict__ OUT,        // (B,C,HW)
                                 int C, int H, int W, int nPix)
{
  int idx = blockIdx.x * blockDim.x + threadIdx.x;
  if (idx >= nPix) return;
  int HW = H * W;
  int b  = idx / HW;
  int hw = idx - b * HW;
  int h  = hw / W, w = hw - h * W;
  int Wp = W + 2, Hp = H + 2;
  int halfC = C >> 1;
  size_t qb = (size_t)b * C * HW + hw;
  size_t pstride = (size_t)Hp * Wp;
  size_t pb = (size_t)b * C * pstride + (size_t)h * Wp + w;

  float s[9];
#pragma unroll
  for (int i = 0; i < 9; ++i) s[i] = 0.0f;

  for (int c = 0; c < C; ++c) {
    float qc = Q[qb + (size_t)c * HW];
    const float* kc = KP + pb + (size_t)c * pstride;
    if (c < halfC) {
#pragma unroll
      for (int kh = 0; kh < 3; ++kh) {
        float bias = rh[c * 3 + kh];
#pragma unroll
        for (int kw = 0; kw < 3; ++kw)
          s[kh * 3 + kw] = fmaf(kc[kh * Wp + kw] + bias, qc, s[kh * 3 + kw]);
      }
    } else {
#pragma unroll
      for (int kh = 0; kh < 3; ++kh)
#pragma unroll
        for (int kw = 0; kw < 3; ++kw)
          s[kh * 3 + kw] = fmaf(kc[kh * Wp + kw] + rw[(c - halfC) * 3 + kw], qc,
                                s[kh * 3 + kw]);
    }
  }

  float mx = s[0];
#pragma unroll
  for (int i = 1; i < 9; ++i) mx = fmaxf(mx, s[i]);
  float sum = 0.0f;
#pragma unroll
  for (int i = 0; i < 9; ++i) { s[i] = __expf(s[i] - mx); sum += s[i]; }
  float inv = 1.0f / sum;
#pragma unroll
  for (int i = 0; i < 9; ++i) s[i] *= inv;

  for (int c = 0; c < C; ++c) {
    const float* vc = VP + pb + (size_t)c * pstride;
    float acc = 0.0f;
#pragma unroll
    for (int kh = 0; kh < 3; ++kh)
#pragma unroll
      for (int kw = 0; kw < 3; ++kw)
        acc = fmaf(s[kh * 3 + kw], vc[kh * Wp + kw], acc);
    OUT[qb + (size_t)c * HW] = acc;
  }
}

// ---------------------------------------------------------------------------
// training-mode BN: one workgroup per channel computes mean / biased variance
// ---------------------------------------------------------------------------
__global__ void bn_stats_kernel(const float* __restrict__ X, float* __restrict__ stats,
                                int C, int HW, int B)
{
  __shared__ float sh0[256];
  __shared__ float sh1[256];
  int c = blockIdx.x;
  float s = 0.0f, s2 = 0.0f;
  for (int b = 0; b < B; ++b) {
    const float* p = X + ((size_t)b * C + c) * HW;
    for (int i = threadIdx.x; i < HW; i += blockDim.x) {
      float v = p[i]; s += v; s2 = fmaf(v, v, s2);
    }
  }
  sh0[threadIdx.x] = s; sh1[threadIdx.x] = s2;
  __syncthreads();
  for (int o = 128; o > 0; o >>= 1) {
    if ((int)threadIdx.x < o) {
      sh0[threadIdx.x] += sh0[threadIdx.x + o];
      sh1[threadIdx.x] += sh1[threadIdx.x + o];
    }
    __syncthreads();
  }
  if (threadIdx.x == 0) {
    float N = (float)(B * HW);
    float mean = sh0[0] / N;
    float var  = sh1[0] / N - mean * mean;   // biased, matches jnp.var
    stats[2 * c]     = mean;
    stats[2 * c + 1] = var;
  }
}

__global__ void bn_relu_kernel(float* __restrict__ X, const float* __restrict__ stats,
                               const float* __restrict__ g, const float* __restrict__ bb,
                               int C, int HW, int n)
{
  int idx = blockIdx.x * blockDim.x + threadIdx.x;
  if (idx >= n) return;
  int c = (idx / HW) % C;
  float m = stats[2 * c], v = stats[2 * c + 1];
  float y = (X[idx] - m) * rsqrtf(v + BN_EPS) * g[c] + bb[c];
  X[idx] = fmaxf(y, 0.0f);
}

__global__ void maxpool2_kernel(const float* __restrict__ X, float* __restrict__ Y,
                                int C, int H, int W, int n)
{
  int idx = blockIdx.x * blockDim.x + threadIdx.x;
  if (idx >= n) return;                       // n = B*C*(H/2)*(W/2)
  int Ho = H >> 1, Wo = W >> 1;
  int wo = idx % Wo;
  int t  = idx / Wo;
  int ho = t % Ho; t /= Ho;
  int c  = t % C;
  int b  = t / C;
  const float* p = X + (((size_t)b * C + c) * H + (size_t)ho * 2) * W + wo * 2;
  float a0 = fmaxf(p[0], p[1]);
  float a1 = fmaxf(p[W], p[W + 1]);
  Y[idx] = fmaxf(a0, a1);
}

// ---------------------------------------------------------------------------
// FC head (tiny: <= 17 MFLOP total) — one thread per output element
// ---------------------------------------------------------------------------
__global__ void fc_kernel(const float* __restrict__ h, const float* __restrict__ Wm,
                          const float* __restrict__ bv, float* __restrict__ out,
                          int fin, int fout, int B, int doRelu)
{
  int idx = blockIdx.x * blockDim.x + threadIdx.x;
  if (idx >= B * fout) return;
  int b = idx / fout;
  int o = idx - b * fout;
  const float* hb = h  + (size_t)b * fin;
  const float* wr = Wm + (size_t)o * fin;
  float acc = bv[o];
  for (int i = 0; i < fin; ++i) acc = fmaf(hb[i], wr[i], acc);
  if (doRelu) acc = fmaxf(acc, 0.0f);
  out[(size_t)b * fout + o] = acc;
}

// ---------------------------------------------------------------------------
// host orchestration
// ---------------------------------------------------------------------------
extern "C" void kernel_launch(void* const* d_in, const int* in_sizes, int n_in,
                              void* d_out, int out_size, void* d_ws, size_t ws_size,
                              hipStream_t stream)
{
  (void)in_sizes; (void)n_in; (void)out_size; (void)ws_size;
  // jax tree-flatten order (dict keys sorted alphabetically):
  //   params.attn[i]: b, g, rh, rw, wk, wq, wv   -> indices i*7 + {0..6}
  //   params.fc[j]:   (W, b)                     -> indices 77 + 2j, 78 + 2j
  //   x                                          -> index 85
  const float* xin = (const float*)d_in[85];

  struct LayerDesc { int cin, cout, H, W, pool; };
  static const LayerDesc L[11] = {
    {3,16,512,512,0},{16,16,512,512,1},{16,32,256,256,0},{32,32,256,256,1},
    {32,64,128,128,0},{64,64,128,128,1},{64,64,64,64,0},{64,64,64,64,1},
    {64,64,32,32,0},{64,64,32,32,1},{64,64,16,16,1}
  };
  static const int FCdim[5] = {4096, 1024, 256, 64, 10};

  size_t off = 0;
  auto carveB = [&](size_t bytes) -> void* {
    void* p = (void*)((char*)d_ws + off);
    off += (bytes + 255) & ~(size_t)255;
    return p;
  };
  const size_t MAXACT = (size_t)2 * 16 * 512 * 512;   // 8.39M floats (largest act)
  const size_t MAXPAD = (size_t)2 * 16 * 514 * 514;   // 8.45M floats (largest padded)
  float* bufs[2];
  bufs[0] = (float*)carveB(MAXACT * 4);
  bufs[1] = (float*)carveB(MAXACT * 4);
  float* Qb = (float*)carveB(MAXACT * 4);
  float* KP = (float*)carveB(MAXPAD * 4);
  float* VP = (float*)carveB(MAXPAD * 4);
  _Float16* XH = (_Float16*)carveB((size_t)2 * 512 * 512 * 32 * 2); // K-major f16 acts
  _Float16* WHq = (_Float16*)carveB(64 * 64 * 2);
  _Float16* WHk = (_Float16*)carveB(64 * 64 * 2);
  _Float16* WHv = (_Float16*)carveB(64 * 64 * 2);
  float* ST = (float*)carveB(256 * 4);
  float* FA = (float*)carveB(2 * 1024 * 4);
  float* FB = (float*)carveB(2 * 1024 * 4);

  const int B = 2;
  const float* src = xin;
  int srcIdx = -1;   // -1: src is the (read-only) network input

  for (int i = 0; i < 11; ++i) {
    int Cin = L[i].cin, Cout = L[i].cout, H = L[i].H, W = L[i].W;
    int HW = H * W, Wp = W + 2, padN = (H + 2) * Wp;
    int Cinp = (Cin <= 32) ? 32 : 64;
    const float* p_b  = (const float*)d_in[i * 7 + 0];
    const float* p_g  = (const float*)d_in[i * 7 + 1];
    const float* p_rh = (const float*)d_in[i * 7 + 2];
    const float* p_rw = (const float*)d_in[i * 7 + 3];
    const float* p_wk = (const float*)d_in[i * 7 + 4];
    const float* p_wq = (const float*)d_in[i * 7 + 5];
    const float* p_wv = (const float*)d_in[i * 7 + 6];

    // f16 conversions: weights (tiny) + activations (K-contiguous, zero padded K)
    int nW = Cout * Cinp;
    cvt_w_kernel<<<(nW + 255) / 256, 256, 0, stream>>>(p_wq, WHq, Cin, Cinp, nW);
    cvt_w_kernel<<<(nW + 255) / 256, 256, 0, stream>>>(p_wk, WHk, Cin, Cinp, nW);
    cvt_w_kernel<<<(nW + 255) / 256, 256, 0, stream>>>(p_wv, WHv, Cin, Cinp, nW);
    int nX = B * HW;
    cvt_x_kernel<<<(nX + 255) / 256, 256, 0, stream>>>(src, XH, Cin, Cinp, HW, nX);

    // zero only the borders of the padded k/v buffers
    int perim = 2 * Wp + 2 * H;
    int nBorder = B * Cout * perim;
    pad_border_kernel<<<(nBorder + 255) / 256, 256, 0, stream>>>(KP, Wp, H, perim, nBorder);
    pad_border_kernel<<<(nBorder + 255) / 256, 256, 0, stream>>>(VP, Wp, H, perim, nBorder);

    // q/k/v 1x1 convs through WMMA (16x64 tile per wave)
    int nWaves = B * (Cout >> 4) * (HW >> 6);
    int blks = (nWaves * 32 + 255) / 256;
    conv1x1_wmma_kernel<<<blks, 256, 0, stream>>>(WHq, XH, Qb, Cinp, Cout, HW, W, 0, Wp, padN, nWaves);
    conv1x1_wmma_kernel<<<blks, 256, 0, stream>>>(WHk, XH, KP, Cinp, Cout, HW, W, 1, Wp, padN, nWaves);
    conv1x1_wmma_kernel<<<blks, 256, 0, stream>>>(WHv, XH, VP, Cinp, Cout, HW, W, 1, Wp, padN, nWaves);

    int dstIdx = (srcIdx == 0) ? 1 : 0;
    float* dst = bufs[dstIdx];
    int nPix = B * HW;
    attention_kernel<<<(nPix + 255) / 256, 256, 0, stream>>>(Qb, KP, VP, p_rh, p_rw,
                                                             dst, Cout, H, W, nPix);

    bn_stats_kernel<<<Cout, 256, 0, stream>>>(dst, ST, Cout, HW, B);
    int nEl = B * Cout * HW;
    bn_relu_kernel<<<(nEl + 255) / 256, 256, 0, stream>>>(dst, ST, p_g, p_b, Cout, HW, nEl);

    if (L[i].pool) {
      int poolIdx = 1 - dstIdx;
      int nOut = B * Cout * (HW >> 2);
      maxpool2_kernel<<<(nOut + 255) / 256, 256, 0, stream>>>(dst, bufs[poolIdx], Cout, H, W, nOut);
      src = bufs[poolIdx]; srcIdx = poolIdx;
    } else {
      src = dst; srcIdx = dstIdx;
    }
  }

  // FC head: (2,4096) -> 1024 -> 256 -> 64 -> 10
  const float* h = src;
  float* fcbuf[2] = {FA, FB};
  for (int j = 0; j < 4; ++j) {
    const float* Wm = (const float*)d_in[77 + j * 2];
    const float* bv = (const float*)d_in[77 + j * 2 + 1];
    int fin = FCdim[j], fout = FCdim[j + 1];
    float* o = (j == 3) ? (float*)d_out : fcbuf[j & 1];
    int n = B * fout;
    fc_kernel<<<(n + 255) / 256, 256, 0, stream>>>(h, Wm, bv, o, fin, fout, B, (j < 3) ? 1 : 0);
    h = o;
  }
}
